// Attention_26525718020753
// MI455X (gfx1250) — matrix operand order repaired and checked by reading, not compiled, not run on hardware
//
#include <hip/hip_runtime.h>

// ---------------------------------------------------------------------------
// Types
// ---------------------------------------------------------------------------
typedef unsigned short u16;
typedef unsigned int   u32;
typedef unsigned long long u64;
typedef __bf16 bf16;
typedef bf16  v16bf __attribute__((ext_vector_type(16)));
typedef bf16  v8bf  __attribute__((ext_vector_type(8)));
typedef float v8f   __attribute__((ext_vector_type(8)));

union BFrag { v16bf v; v8bf h[2]; };

__device__ __forceinline__ u16 f32_to_bf16_rne(float f) {
  u32 u = __builtin_bit_cast(u32, f);
  u32 r = u + 0x7FFFu + ((u >> 16) & 1u);   // round-to-nearest-even
  return (u16)(r >> 16);
}
__device__ __forceinline__ float bf16_to_f32(u16 h) {
  return __builtin_bit_cast(float, (u32)h << 16);
}

// Problem constants (match reference)
#define NT   128
#define NB   256
#define NC   512
#define NH   512
#define NL   26
#define NCLS 6736

// ---------------------------------------------------------------------------
// f32 -> bf16 conversion
// ---------------------------------------------------------------------------
__global__ void k_f32_to_bf16(const float* __restrict__ in, u16* __restrict__ out,
                              size_t n) {
  size_t i = (size_t)blockIdx.x * blockDim.x + threadIdx.x;
  if (i < n) out[i] = f32_to_bf16_rne(in[i]);
}

__global__ void k_zero_h(float* __restrict__ hf, u16* __restrict__ hb, int n) {
  int i = blockIdx.x * blockDim.x + threadIdx.x;
  if (i < n) { hf[i] = 0.f; hb[i] = 0; }
}

// ---------------------------------------------------------------------------
// Tiled bf16 WMMA GEMM: C[M,N] = A[M,K] @ Bt[N,K]^T (+ bias)
// A, Bt bf16 row-major. Output f32 (Cf) or bf16 (Cb) per outBf.
// Block = 256 threads = 8 waves; block tile 64(M) x 64(N); K step 32.
// Double-buffered LDS staging via GLOBAL_LOAD_ASYNC_TO_LDS_B128 (ASYNCcnt):
// each wave issues exactly 2 async loads per K-tile (A slice + B slice), so
// s_wait_asynccnt 2 = "current tile landed, next tile in flight".
// N tail (generator: 6736 % 64 == 16): B row index clamped to N-1 (keeps the
// per-wave async count EXEC-independent); garbage affects only output columns
// >= N, which are per-column independent and masked at the store.
// ---------------------------------------------------------------------------
__global__ __launch_bounds__(256) void k_gemm_bf16_wmma(
    const u16* __restrict__ A, const u16* __restrict__ Bt,
    const float* __restrict__ bias,
    float* __restrict__ Cf, u16* __restrict__ Cb,
    int M, int N, int K, int outBf) {
  __shared__ u16 lA[2][64 * 32];
  __shared__ u16 lB[2][64 * 32];

  const int tid  = threadIdx.x;
  const int wave = tid >> 5;
  const int lane = tid & 31;
  const int m0 = blockIdx.y * 64;
  const int n0 = blockIdx.x * 64;
  const int mIdx = wave >> 1;      // 0..3
  const int nIdx = wave & 1;       // 0..1

  const int ldrow = tid >> 2;           // 0..63
  const int ldcol = (tid & 3) * 8;      // 0,8,16,24

  // Staging source pointers (B row clamped: OOB rows feed only discarded cols)
  const int brow  = n0 + ldrow;
  const int browc = (brow < N) ? brow : (N - 1);
  const u16* gA = &A[(size_t)(m0 + ldrow) * K + ldcol];
  const u16* gB = &Bt[(size_t)browc * K + ldcol];

  v8f acc0 = {};
  v8f acc1 = {};

  const int l15 = lane & 15;
  const int hiA = (lane < 16) ? 0 : 8;    // A frag: lanes<16 K{0..7,16..23}, else K{8..15,24..31}
  const int hiB = (lane < 16) ? 0 : 8;    // B frag: lanes<16 K{0..7,16..23}, else K{8..15,24..31}

  // Prologue: async-stage tile 0 into buffer 0
  {
    u32 la = (u32)(u64)&lA[0][ldrow * 32 + ldcol];
    u32 lb = (u32)(u64)&lB[0][ldrow * 32 + ldcol];
    asm volatile("global_load_async_to_lds_b128 %0, %1, off"
                 :: "v"(la), "v"((u64)gA) : "memory");
    asm volatile("global_load_async_to_lds_b128 %0, %1, off"
                 :: "v"(lb), "v"((u64)gB) : "memory");
  }

  for (int k0 = 0; k0 < K; k0 += 32) {
    const int buf = (k0 >> 5) & 1;
    if (k0 + 32 < K) {
      // Prefetch next tile into the other buffer (its last readers barriered
      // out at the end of the previous iteration).
      const int nb = buf ^ 1;
      u32 la = (u32)(u64)&lA[nb][ldrow * 32 + ldcol];
      u32 lb = (u32)(u64)&lB[nb][ldrow * 32 + ldcol];
      asm volatile("global_load_async_to_lds_b128 %0, %1, off"
                   :: "v"(la), "v"((u64)(gA + k0 + 32)) : "memory");
      asm volatile("global_load_async_to_lds_b128 %0, %1, off"
                   :: "v"(lb), "v"((u64)(gB + k0 + 32)) : "memory");
      // Wait for the CURRENT tile (2 older async loads retire in order)
      asm volatile("s_wait_asynccnt 0x2" ::: "memory");
    } else {
      asm volatile("s_wait_asynccnt 0x0" ::: "memory");
    }
    __syncthreads();   // every wave's slice of tile k0 is in LDS

    // A fragment (16x32, M = mIdx*16 + l15)
    BFrag a;
    a.h[0] = *(v8bf*)&lA[buf][(mIdx * 16 + l15) * 32 + hiA];
    a.h[1] = *(v8bf*)&lA[buf][(mIdx * 16 + l15) * 32 + hiA + 16];
    // B fragments (32x16 each, N column = lane index per ISA layout)
    BFrag b0, b1;
    b0.h[0] = *(v8bf*)&lB[buf][(nIdx * 32 + l15) * 32 + hiB];
    b0.h[1] = *(v8bf*)&lB[buf][(nIdx * 32 + l15) * 32 + hiB + 16];
    b1.h[0] = *(v8bf*)&lB[buf][(nIdx * 32 + 16 + l15) * 32 + hiB];
    b1.h[1] = *(v8bf*)&lB[buf][(nIdx * 32 + 16 + l15) * 32 + hiB + 16];

    acc0 = __builtin_amdgcn_wmma_f32_16x16x32_bf16(false, a.v, false, b0.v,
                                                   (short)0, acc0, false, false);
    acc1 = __builtin_amdgcn_wmma_f32_16x16x32_bf16(false, a.v, false, b1.v,
                                                   (short)0, acc1, false, false);
    __syncthreads();   // all waves done reading buf before it is re-filled
  }

  // Store: C/D layout — VGPR v holds M = base+v; lanes<16 rows 0..7, lanes>=16 rows 8..15
  const int mbase = m0 + mIdx * 16 + ((lane < 16) ? 0 : 8);
  for (int nt = 0; nt < 2; nt++) {
    v8f acc = nt ? acc1 : acc0;
    int n = n0 + nIdx * 32 + nt * 16 + l15;
    if (n >= N) continue;
    float bv = bias ? bias[n] : 0.f;
    for (int v = 0; v < 8; v++) {
      int m = mbase + v;
      float val = acc[v] + bv;
      size_t ci = (size_t)m * N + n;
      if (outBf) Cb[ci] = f32_to_bf16_rne(val);
      else       Cf[ci] = val;
    }
  }
}

// ---------------------------------------------------------------------------
// Attention score: e[b,t] = sum_h tanh(fproj[t,b,h] + hp[b,h]) * wscore[h]
// One wave32 per (t,b); lane covers 16 of H=512.
// ---------------------------------------------------------------------------
__global__ __launch_bounds__(256) void k_attn_score(
    const u16* __restrict__ fprojb, const float* __restrict__ hp,
    const float* __restrict__ wscore, float* __restrict__ e) {
  const int wave = threadIdx.x >> 5;
  const int lane = threadIdx.x & 31;
  const int pair = blockIdx.x * 8 + wave;       // flat (t,b)
  const int t = pair / NB;
  const int b = pair % NB;
  const u16*   fp  = fprojb + ((size_t)t * NB + b) * NH + lane * 16;
  const float* hpb = hp + (size_t)b * NH + lane * 16;
  const float* ws  = wscore + lane * 16;

  v8bf f0 = *(const v8bf*)fp;
  v8bf f1 = *(const v8bf*)(fp + 8);
  float s = 0.f;
  for (int i = 0; i < 8; i++) s += tanhf((float)f0[i] + hpb[i]) * ws[i];
  for (int i = 0; i < 8; i++) s += tanhf((float)f1[i] + hpb[8 + i]) * ws[8 + i];
  for (int off = 16; off >= 1; off >>= 1) s += __shfl_down(s, off, 32);
  if (lane == 0) e[(size_t)b * NT + t] = s;
}

// ---------------------------------------------------------------------------
// Softmax over T (=128) per batch element
// ---------------------------------------------------------------------------
__global__ __launch_bounds__(128) void k_softmax_T(const float* __restrict__ e,
                                                   float* __restrict__ alpha) {
  __shared__ float red[128];
  const int b = blockIdx.x, t = threadIdx.x;
  float v = e[(size_t)b * NT + t];
  red[t] = v; __syncthreads();
  for (int s = 64; s > 0; s >>= 1) {
    if (t < s) red[t] = fmaxf(red[t], red[t + s]);
    __syncthreads();
  }
  float mx = red[0]; __syncthreads();
  float ex = expf(v - mx);
  red[t] = ex; __syncthreads();
  for (int s = 64; s > 0; s >>= 1) {
    if (t < s) red[t] += red[t + s];
    __syncthreads();
  }
  alpha[(size_t)b * NT + t] = ex / red[0];
}

// ---------------------------------------------------------------------------
// context[b,c] = sum_t alpha[b,t] * feats[t,b,c]   (bf16 streaming, bf16 out)
// One block per b; thread handles 2 channels (one dword load per t).
// ---------------------------------------------------------------------------
__global__ __launch_bounds__(256) void k_context(
    const float* __restrict__ alpha, const u16* __restrict__ featsb,
    u16* __restrict__ ctxb) {
  const int b = blockIdx.x;
  const int c = threadIdx.x * 2;
  float a0 = 0.f, a1 = 0.f;
  for (int t = 0; t < NT; t++) {
    float al = alpha[(size_t)b * NT + t];
    u32 u = *(const u32*)&featsb[((size_t)t * NB + b) * NC + c];
    a0 += al * bf16_to_f32((u16)u);
    a1 += al * bf16_to_f32((u16)(u >> 16));
  }
  ctxb[(size_t)b * NC + c]     = f32_to_bf16_rne(a0);
  ctxb[(size_t)b * NC + c + 1] = f32_to_bf16_rne(a1);
}

// ---------------------------------------------------------------------------
// Fused GRU gates + state update + output-history write (b*L + step order)
// ---------------------------------------------------------------------------
__global__ __launch_bounds__(256) void k_gru(
    const float* __restrict__ gi, const float* __restrict__ gh,
    float* __restrict__ hf, u16* __restrict__ hb, u16* __restrict__ hsb,
    int step) {
  const int idx = blockIdx.x * blockDim.x + threadIdx.x;   // B*H
  const int b = idx / NH, j = idx % NH;
  const size_t base = (size_t)b * 3 * NH + j;
  float ir = gi[base], iz = gi[base + NH], in_ = gi[base + 2 * NH];
  float hr = gh[base], hz = gh[base + NH], hn  = gh[base + 2 * NH];
  float r = 1.f / (1.f + expf(-(ir + hr)));
  float z = 1.f / (1.f + expf(-(iz + hz)));
  float n = tanhf(in_ + r * hn);
  float hnew = (1.f - z) * n + z * hf[idx];
  hf[idx] = hnew;
  u16 hv = f32_to_bf16_rne(hnew);
  hb[idx] = hv;
  hsb[((size_t)b * NL + step) * NH + j] = hv;
}

// ---------------------------------------------------------------------------
// Host launch
// ---------------------------------------------------------------------------
extern "C" void kernel_launch(void* const* d_in, const int* in_sizes, int n_in,
                              void* d_out, int out_size, void* d_ws, size_t ws_size,
                              hipStream_t stream) {
  const float* feats   = (const float*)d_in[0];
  // d_in[1] = text_length (uniform NL, loop count is static)
  const float* W_i2h   = (const float*)d_in[2];
  const float* W_h2h   = (const float*)d_in[3];
  const float* b_h2h   = (const float*)d_in[4];
  const float* W_score = (const float*)d_in[5];
  const float* W_ih    = (const float*)d_in[6];
  const float* W_hh    = (const float*)d_in[7];
  const float* b_ih    = (const float*)d_in[8];
  const float* b_hh    = (const float*)d_in[9];
  const float* W_gen   = (const float*)d_in[10];
  const float* b_gen   = (const float*)d_in[11];

  char* base = (char*)d_ws;
  size_t off = 0;
  auto alloc = [&](size_t bytes) -> char* {
    char* p = base + off;
    off += (bytes + 255) & ~(size_t)255;
    return p;
  };
  u16*   featsb = (u16*)  alloc((size_t)NT * NB * NC * 2);
  u16*   fprojb = (u16*)  alloc((size_t)NT * NB * NH * 2);
  u16*   wi2hb  = (u16*)  alloc((size_t)NH * NC * 2);
  u16*   wh2hb  = (u16*)  alloc((size_t)NH * NH * 2);
  u16*   wihb   = (u16*)  alloc((size_t)3 * NH * NC * 2);
  u16*   whhb   = (u16*)  alloc((size_t)3 * NH * NH * 2);
  u16*   wgenb  = (u16*)  alloc((size_t)NCLS * NH * 2);
  float* hp     = (float*)alloc((size_t)NB * NH * 4);
  float* e      = (float*)alloc((size_t)NB * NT * 4);
  float* alpha  = (float*)alloc((size_t)NB * NT * 4);
  u16*   ctxb   = (u16*)  alloc((size_t)NB * NC * 2);
  float* gi     = (float*)alloc((size_t)NB * 3 * NH * 4);
  float* gh     = (float*)alloc((size_t)NB * 3 * NH * 4);
  float* hf     = (float*)alloc((size_t)NB * NH * 4);
  u16*   hb     = (u16*)  alloc((size_t)NB * NH * 2);
  u16*   hsb    = (u16*)  alloc((size_t)NB * NL * NH * 2);
  (void)ws_size; (void)n_in; (void)in_sizes; (void)out_size;

  auto conv = [&](const float* src, u16* dst, size_t n) {
    k_f32_to_bf16<<<(unsigned)((n + 255) / 256), 256, 0, stream>>>(src, dst, n);
  };
  conv(feats, featsb, (size_t)NT * NB * NC);
  conv(W_i2h, wi2hb, (size_t)NH * NC);
  conv(W_h2h, wh2hb, (size_t)NH * NH);
  conv(W_ih,  wihb,  (size_t)3 * NH * NC);
  conv(W_hh,  whhb,  (size_t)3 * NH * NH);
  conv(W_gen, wgenb, (size_t)NCLS * NH);
  k_zero_h<<<(NB * NH) / 256, 256, 0, stream>>>(hf, hb, NB * NH);

  // feats_proj = feats @ W_i2h^T  -> bf16 [T*B, H]
  k_gemm_bf16_wmma<<<dim3(NH / 64, (NT * NB) / 64), 256, 0, stream>>>(
      featsb, wi2hb, nullptr, nullptr, fprojb, NT * NB, NH, NC, 1);

  for (int step = 0; step < NL; step++) {
    // hp = h @ W_h2h^T + b_h2h
    k_gemm_bf16_wmma<<<dim3(NH / 64, NB / 64), 256, 0, stream>>>(
        hb, wh2hb, b_h2h, hp, nullptr, NB, NH, NH, 0);
    k_attn_score<<<(NT * NB) / 8, 256, 0, stream>>>(fprojb, hp, W_score, e);
    k_softmax_T<<<NB, 128, 0, stream>>>(e, alpha);
    k_context<<<NB, 256, 0, stream>>>(alpha, featsb, ctxb);
    // gi = context @ W_ih^T + b_ih ; gh = h @ W_hh^T + b_hh
    k_gemm_bf16_wmma<<<dim3((3 * NH) / 64, NB / 64), 256, 0, stream>>>(
        ctxb, wihb, b_ih, gi, nullptr, NB, 3 * NH, NC, 0);
    k_gemm_bf16_wmma<<<dim3((3 * NH) / 64, NB / 64), 256, 0, stream>>>(
        hb, whhb, b_hh, gh, nullptr, NB, 3 * NH, NH, 0);
    k_gru<<<(NB * NH) / 256, 256, 0, stream>>>(gi, gh, hf, hb, hsb, step);
  }

  // probs = out_h @ W_gen^T + b_gen   [B*L, NCLS] f32 -> d_out
  k_gemm_bf16_wmma<<<dim3((NCLS + 63) / 64, (NB * NL) / 64), 256, 0, stream>>>(
      hsb, wgenb, b_gen, (float*)d_out, nullptr, NB * NL, NCLS, NH, 0);
}
